// RotatedModule_8323646620479
// MI455X (gfx1250) — matrix operand order repaired and checked by reading, not compiled, not run on hardware
//
#include <hip/hip_runtime.h>

// ---------------------------------------------------------------------------
// RotatedModule: out = (blockHadamardRotate(inp)) @ W.T + b
//   M = B*S = 8192, N = K = 4096, k = 16 segments of 256
//   1) rotate_fwht_kernel : fp32 FWHT(256) + 16x16 had mix -> bf16 A' (M x K)
//   2) w_bf16_kernel      : W -> bf16 (row-major n,k == B operand W^T)
//   3) gemm_bf16_wmma     : v_wmma_f32_16x16x32_bf16, 128x128 tile, BK=32,
//      double-buffered LDS fed by GLOBAL_LOAD_ASYNC_TO_LDS_B128 (ASYNCcnt).
// Workspace: A' = 64 MiB @ ws+0, Wb = 32 MiB @ ws+64MiB  (96 MiB total).
// ---------------------------------------------------------------------------

typedef __attribute__((ext_vector_type(16))) __bf16 v16bf;
typedef __attribute__((ext_vector_type(8)))  __bf16 v8bf;
typedef __attribute__((ext_vector_type(8)))  float  v8f;
typedef __attribute__((ext_vector_type(4)))  int    v4i;

// Builtin parameter types (from hipcc diagnostic): arg0 = v4i in AS1 (global),
// arg1 = LDS destination (AS3), both non-const.
typedef __attribute__((address_space(1))) v4i* gas_v4i;
typedef __attribute__((address_space(3))) v4i* las_v4i;

#define MDIM 8192
#define NDIM 4096
#define KDIM 4096

#if __has_builtin(__builtin_amdgcn_global_load_async_to_lds_b128)
#define HAS_ASYNC_LDS 1
#else
#define HAS_ASYNC_LDS 0
#endif

__device__ __forceinline__ void wait_async0() {
#if __has_builtin(__builtin_amdgcn_s_wait_asynccnt)
    __builtin_amdgcn_s_wait_asynccnt(0);
#else
    asm volatile("s_wait_asynccnt 0x0" ::: "memory");
#endif
}

// ---------------- Kernel 1: per-row FWHT + had_mat mix -> bf16 --------------
__global__ __launch_bounds__(256) void rotate_fwht_kernel(
    const float* __restrict__ inp, const float* __restrict__ had,
    __bf16* __restrict__ out)
{
    __shared__ float buf[4096];
    __shared__ float hmat[256];
    const int t = threadIdx.x;
    const long row = blockIdx.x;

    const float4* s4 = (const float4*)(inp + row * 4096);
    float4* b4 = (float4*)buf;
#pragma unroll
    for (int i = 0; i < 4; ++i) b4[t + 256 * i] = s4[t + 256 * i];
    hmat[t] = had[t];
    __syncthreads();

    // FWHT over 16 segments of 256 (Sylvester order, matches reference)
    for (int lg = 0; lg < 8; ++lg) {
        const int h = 1 << lg;
#pragma unroll
        for (int r = 0; r < 8; ++r) {
            const int p = t + 256 * r;     // 2048 butterflies / stage
            const int seg = p >> 7;
            const int j = p & 127;
            const int i1 = seg * 256 + ((j >> lg) << (lg + 1)) + (j & (h - 1));
            const float a = buf[i1];
            const float c = buf[i1 + h];
            buf[i1]     = a + c;
            buf[i1 + h] = a - c;
        }
        __syncthreads();
    }

    // 16x16 had_mat mix across segments at column d = t, scale 1/sqrt(4096)
    float x[16];
#pragma unroll
    for (int j = 0; j < 16; ++j) x[j] = buf[j * 256 + t];
    const float scale = 0.015625f;  // 1/64
    __bf16* dst = out + row * 4096 + t;
#pragma unroll
    for (int i = 0; i < 16; ++i) {
        float y = 0.f;
#pragma unroll
        for (int j = 0; j < 16; ++j) y += hmat[i * 16 + j] * x[j];
        dst[i * 256] = (__bf16)(y * scale);
    }
}

// ---------------- Kernel 2: W fp32 -> bf16 ----------------------------------
__global__ __launch_bounds__(256) void w_bf16_kernel(
    const float* __restrict__ W, __bf16* __restrict__ Wb)
{
    const long i = ((long)blockIdx.x * 256 + threadIdx.x) * 8;
    const float4 a = *(const float4*)(W + i);
    const float4 c = *(const float4*)(W + i + 4);
    v8bf o;
    o[0] = (__bf16)a.x; o[1] = (__bf16)a.y; o[2] = (__bf16)a.z; o[3] = (__bf16)a.w;
    o[4] = (__bf16)c.x; o[5] = (__bf16)c.y; o[6] = (__bf16)c.z; o[7] = (__bf16)c.w;
    *(v8bf*)(Wb + i) = o;
}

// ---------------- Kernel 3: bf16 WMMA GEMM  C = A' * Wb^T + b ---------------
// Block tile 128(M) x 128(N), BK = 32.  256 threads = 8 waves in a 2x4 grid;
// each wave owns a 64x32 sub-tile = 4x2 WMMA 16x16 accumulators.
__global__ __launch_bounds__(256) void gemm_bf16_wmma_kernel(
    const __bf16* __restrict__ A, const __bf16* __restrict__ B,
    const float* __restrict__ bias, float* __restrict__ C)
{
    // Tiles stored k-contiguous: row (m or n) * 32 bf16 = 4 uint4 chunks.
    __shared__ uint4 As[2][512];   // 128 x 32 bf16 = 8 KB each buffer
    __shared__ uint4 Bs[2][512];

    const int tid  = threadIdx.x;
    const int lane = tid & 31;
    const int wave = tid >> 5;
    const int wm   = wave >> 2;      // 0..1  (M direction)
    const int wn   = wave & 3;       // 0..3  (N direction)
    const int l15  = lane & 15;
    const int lh   = lane >> 4;      // lane half
    const int m0   = blockIdx.y * 128;
    const int n0   = blockIdx.x * 128;

    // Each thread moves 2 x 16B for A and 2 x 16B for B per k-step.
    const int ar0 = tid >> 2,         ac0 = tid & 3;         // rows 0..63
    const int ar1 = (tid + 256) >> 2, ac1 = tid & 3;         // rows 64..127
    const uint4* gA = (const uint4*)A;
    const uint4* gB = (const uint4*)B;
    const long strideU4 = KDIM / 8;  // 512 uint4 per row

#if HAS_ASYNC_LDS
    // Direct global -> LDS copy on the async path (ASYNCcnt), no staging VGPRs.
    auto tile_copy = [&](int kt, int bsel) {
        const long ko = (long)kt * 4;
        __builtin_amdgcn_global_load_async_to_lds_b128(
            (gas_v4i)(gA + (long)(m0 + ar0) * strideU4 + ko + ac0),
            (las_v4i)&As[bsel][ar0 * 4 + ac0], 0, 0);
        __builtin_amdgcn_global_load_async_to_lds_b128(
            (gas_v4i)(gA + (long)(m0 + ar1) * strideU4 + ko + ac1),
            (las_v4i)&As[bsel][ar1 * 4 + ac1], 0, 0);
        __builtin_amdgcn_global_load_async_to_lds_b128(
            (gas_v4i)(gB + (long)(n0 + ar0) * strideU4 + ko + ac0),
            (las_v4i)&Bs[bsel][ar0 * 4 + ac0], 0, 0);
        __builtin_amdgcn_global_load_async_to_lds_b128(
            (gas_v4i)(gB + (long)(n0 + ar1) * strideU4 + ko + ac1),
            (las_v4i)&Bs[bsel][ar1 * 4 + ac1], 0, 0);
    };
#else
    uint4 ra0, ra1, rb0, rb1;
    auto gload = [&](int kt) {
        const long ko = (long)kt * 4;
        ra0 = gA[(long)(m0 + ar0) * strideU4 + ko + ac0];
        ra1 = gA[(long)(m0 + ar1) * strideU4 + ko + ac1];
        rb0 = gB[(long)(n0 + ar0) * strideU4 + ko + ac0];
        rb1 = gB[(long)(n0 + ar1) * strideU4 + ko + ac1];
    };
    auto lstore = [&](int bsel) {
        As[bsel][ar0 * 4 + ac0] = ra0;
        As[bsel][ar1 * 4 + ac1] = ra1;
        Bs[bsel][ar0 * 4 + ac0] = rb0;
        Bs[bsel][ar1 * 4 + ac1] = rb1;
    };
#endif

    v8f acc[4][2];
#pragma unroll
    for (int i = 0; i < 4; ++i)
#pragma unroll
        for (int j = 0; j < 2; ++j) acc[i][j] = {};

#if HAS_ASYNC_LDS
    tile_copy(0, 0);
    wait_async0();
    __syncthreads();
#else
    gload(0);
    lstore(0);
    __syncthreads();
#endif

    const int KT = KDIM / 32;   // 128 k-steps
    for (int kt = 0; kt < KT; ++kt) {
        const int cur = kt & 1;
#if HAS_ASYNC_LDS
        if (kt + 1 < KT) tile_copy(kt + 1, cur ^ 1);   // overlaps with WMMA below
#else
        if (kt + 1 < KT) gload(kt + 1);
#endif

        // A fragment (16-bit A 16x32 layout): lane m = l15, half lh;
        //   regs 0..3 hold K = 0..7 (+8 if lh), regs 4..7 hold K = 16..23 (+8 if lh)
        v16bf af[4];
#pragma unroll
        for (int tm = 0; tm < 4; ++tm) {
            const int m = wm * 64 + tm * 16 + l15;
            v8bf lo = *(const v8bf*)&As[cur][m * 4 + lh];
            v8bf hi = *(const v8bf*)&As[cur][m * 4 + 2 + lh];
            af[tm] = __builtin_shufflevector(lo, hi,
                     0,1,2,3,4,5,6,7,8,9,10,11,12,13,14,15);
        }
        // B fragment (bf16 B 32x16): lane n = l15; regs 0..3 K = 0..7 (+8 if lh), regs 4..7 K = 16..23 (+8 if lh)
        v16bf bfr[2];
#pragma unroll
        for (int tn = 0; tn < 2; ++tn) {
            const int n = wn * 32 + tn * 16 + l15;
            v8bf lo = *(const v8bf*)&Bs[cur][n * 4 + lh];
            v8bf hi = *(const v8bf*)&Bs[cur][n * 4 + 2 + lh];
            bfr[tn] = __builtin_shufflevector(lo, hi,
                      0,1,2,3,4,5,6,7,8,9,10,11,12,13,14,15);
        }

#pragma unroll
        for (int tm = 0; tm < 4; ++tm)
#pragma unroll
            for (int tn = 0; tn < 2; ++tn)
                acc[tm][tn] = __builtin_amdgcn_wmma_f32_16x16x32_bf16(
                    false, af[tm], false, bfr[tn],
                    (short)0, acc[tm][tn], false, false);

#if HAS_ASYNC_LDS
        if (kt + 1 < KT) wait_async0();   // copy has overlapped the 8 WMMAs
#else
        if (kt + 1 < KT) lstore((kt + 1) & 1);
#endif
        __syncthreads();
    }

    // Epilogue: C/D 16x16 f32 layout -> reg r holds row (r + 8*lh), col l15.
#pragma unroll
    for (int tn = 0; tn < 2; ++tn) {
        const int n = n0 + wn * 32 + tn * 16 + l15;
        const float bv = bias[n];
#pragma unroll
        for (int tm = 0; tm < 4; ++tm) {
            const int mbase = m0 + wm * 64 + tm * 16 + 8 * lh;
#pragma unroll
            for (int r = 0; r < 8; ++r)
                C[(long)(mbase + r) * NDIM + n] = acc[tm][tn][r] + bv;
        }
    }
}

// ---------------------------------------------------------------------------
extern "C" void kernel_launch(void* const* d_in, const int* in_sizes, int n_in,
                              void* d_out, int out_size, void* d_ws, size_t ws_size,
                              hipStream_t stream) {
    (void)in_sizes; (void)n_in; (void)out_size; (void)ws_size;
    const float* inp  = (const float*)d_in[0];   // (4,2048,4096) f32
    const float* had  = (const float*)d_in[1];   // (16,16) f32
    const float* W    = (const float*)d_in[2];   // (4096,4096) f32
    const float* bias = (const float*)d_in[3];   // (4096,) f32
    float* out = (float*)d_out;

    __bf16* Aprime = (__bf16*)d_ws;                                       // 64 MiB
    __bf16* Wb     = (__bf16*)((char*)d_ws + (size_t)MDIM * KDIM * 2);    // 32 MiB

    w_bf16_kernel<<<(NDIM * (long)KDIM) / (8 * 256), 256, 0, stream>>>(W, Wb);
    rotate_fwht_kernel<<<MDIM, 256, 0, stream>>>(inp, had, Aprime);
    gemm_bf16_wmma_kernel<<<dim3(NDIM / 128, MDIM / 128), 256, 0, stream>>>(
        Aprime, Wb, bias, out);
}
